// GPTSambaMoDFFN_57312043598493
// MI455X (gfx1250) — compile-verified, matrix operand order checked by reading, not run on hardware
//
#include <hip/hip_runtime.h>

typedef __attribute__((ext_vector_type(16))) __bf16 v16bf;
typedef __attribute__((ext_vector_type(8)))  __bf16 v8bf;
typedef __attribute__((ext_vector_type(4)))  __bf16 v4bf;
typedef __attribute__((ext_vector_type(8)))  float  v8f;

// LDS tile geometry: 128 rows x 32 K-elements, padded to 40 elements/row
// (80 B stride -> conflict-free 16 B fragment reads across 16 lanes).
#define BM   128
#define BK   32
#define LDSS 40                      // row stride in elements
#define TILE_ELEMS (BM * LDSS)       // 5120 elements = 10240 bytes per buffer

__device__ __forceinline__ v16bf combine_frag(v8bf lo, v8bf hi) {
    return __builtin_shufflevector(lo, hi, 0,1,2,3,4,5,6,7,8,9,10,11,12,13,14,15);
}

// ---------------------------------------------------------------------------
// Core GEMM tile engine: block tile 128(M) x 128(N), wave tile 64x32,
// K stepped by 32. A/B K-slabs staged to LDS with async global->LDS copies
// (ASYNCcnt), double-buffered so the async engine overlaps WMMA compute.
// ---------------------------------------------------------------------------
__device__ __forceinline__ void gemm_core(const __bf16* __restrict__ Abase,
                                          const __bf16* __restrict__ Bbase,
                                          int K, int rows0, int cols0,
                                          v8f acc[4][2]) {
    __shared__ __bf16 As[2][TILE_ELEMS];
    __shared__ __bf16 Bs[2][TILE_ELEMS];

    const int tid   = threadIdx.x;
    const int lane  = tid & 31;
    const int wave  = tid >> 5;
    const int waveM = wave >> 2;          // 0..1 -> 64-row slice
    const int waveN = wave & 3;           // 0..3 -> 32-col slice
    const int lr    = lane & 15;
    const int kh    = (lane >> 4) << 3;   // +8 K for upper half-wave

    // --- async copy plan: 1024 16-byte chunks per K-slab pair, 4 per thread
    uint64_t ga[4];                        // global source addresses (k0 = 0)
    uint32_t ls[4];                        // LDS dest offsets (buffer 0)
#pragma unroll
    for (int j = 0; j < 4; ++j) {
        int  c   = tid + j * 256;
        bool isA = c < 512;
        int  cc  = isA ? c : c - 512;
        int  row = cc >> 2;               // 0..127
        int  kc  = cc & 3;                // 16B chunk within the 32-K row
        const __bf16* gp = isA
            ? (Abase + (size_t)(rows0 + row) * K + kc * 8)
            : (Bbase + (size_t)(cols0 + row) * K + kc * 8);
        ga[j] = (uint64_t)(uintptr_t)gp;
        uint32_t lbase = isA ? (uint32_t)(uintptr_t)(&As[0][0])
                             : (uint32_t)(uintptr_t)(&Bs[0][0]);
        ls[j] = lbase + (uint32_t)(row * (LDSS * 2) + kc * 16);
    }

    const int nsteps = K / BK;

    // prologue: fill buffer 0
#pragma unroll
    for (int j = 0; j < 4; ++j)
        asm volatile("global_load_async_to_lds_b128 %0, %1, off"
                     :: "v"(ls[j]), "v"(ga[j]) : "memory");

    for (int s = 0; s < nsteps; ++s) {
        // own slab for buffer s&1 has landed in LDS
        asm volatile("s_wait_asynccnt 0x0" ::: "memory");
        // everyone's slab landed; everyone done reading buffer (s+1)&1
        __syncthreads();

        if (s + 1 < nsteps) {
            uint32_t boff = (uint32_t)(((s + 1) & 1) * (TILE_ELEMS * 2));
            uint64_t koff = (uint64_t)(s + 1) * (BK * 2);
#pragma unroll
            for (int j = 0; j < 4; ++j)
                asm volatile("global_load_async_to_lds_b128 %0, %1, off"
                             :: "v"(ls[j] + boff), "v"(ga[j] + koff) : "memory");
        }

        const __bf16* Asb = &As[s & 1][0];
        const __bf16* Bsb = &Bs[s & 1][0];

        v16bf a[4], b[2];
#pragma unroll
        for (int mi = 0; mi < 4; ++mi) {
            const __bf16* p = Asb + (waveM * 64 + mi * 16 + lr) * LDSS + kh;
            a[mi] = combine_frag(*(const v8bf*)p, *(const v8bf*)(p + 16));
        }
#pragma unroll
        for (int ni = 0; ni < 2; ++ni) {
            const __bf16* p = Bsb + (waveN * 32 + ni * 16 + lr) * LDSS + kh;
            b[ni] = combine_frag(*(const v8bf*)p, *(const v8bf*)(p + 16));
        }
#pragma unroll
        for (int mi = 0; mi < 4; ++mi)
#pragma unroll
            for (int ni = 0; ni < 2; ++ni)
                acc[mi][ni] = __builtin_amdgcn_wmma_f32_16x16x32_bf16(
                    false, a[mi], false, b[ni], (short)0, acc[mi][ni],
                    false, false);
    }
}

// ---------------------------------------------------------------------------
// Kernel 0: zero the compaction counter
// ---------------------------------------------------------------------------
__global__ void zero_counter_kernel(int* counter) {
    if (threadIdx.x == 0) *counter = 0;
}

// ---------------------------------------------------------------------------
// Kernel 1: fp32 -> bf16 weight conversion (vectorized 4-wide)
// ---------------------------------------------------------------------------
__global__ void cvt_bf16_kernel(const float* __restrict__ in,
                                __bf16* __restrict__ out, size_t n) {
    size_t i = ((size_t)blockIdx.x * blockDim.x + threadIdx.x) * 4;
    if (i + 3 < n) {
        float4 v = *(const float4*)(in + i);
        v4bf o;
        o[0] = (__bf16)v.x; o[1] = (__bf16)v.y;
        o[2] = (__bf16)v.z; o[3] = (__bf16)v.w;
        *(v4bf*)(out + i) = o;
    } else {
        for (size_t j = i; j < n; ++j) out[j] = (__bf16)in[j];
    }
}

// ---------------------------------------------------------------------------
// Kernel 2: router + RMS-norm + token compaction + residual copy.
// ---------------------------------------------------------------------------
__global__ __launch_bounds__(256)
void router_rms_kernel(const float* __restrict__ x,
                       const float* __restrict__ w_router,
                       float* __restrict__ out,
                       __bf16* __restrict__ xn,
                       int* __restrict__ idx,
                       float* __restrict__ maskv,
                       int* __restrict__ counter,
                       int C) {
    const int t   = blockIdx.x;
    const int tid = threadIdx.x;
    const float* xr = x + (size_t)t * C;

    float ss = 0.f, dd = 0.f;
    for (int i = tid; i < C; i += blockDim.x) {
        float v = xr[i];
        ss += v * v;
        dd += v * w_router[i];
    }
    __shared__ float s1[256], s2[256];
    s1[tid] = ss; s2[tid] = dd;
    __syncthreads();
    for (int st = 128; st > 0; st >>= 1) {
        if (tid < st) { s1[tid] += s1[tid + st]; s2[tid] += s2[tid + st]; }
        __syncthreads();
    }
    __shared__ float sh_rstd;
    __shared__ int   sh_slot;
    if (tid == 0) {
        float ms    = s1[0] / (float)C;
        float rstd  = rsqrtf(ms + 1.1920929e-07f);
        float logit = s2[0];
        float prob  = 1.f / (1.f + __expf(-logit));
        float hard  = (prob > 0.5f) ? 1.f : 0.f;
        float mk    = (hard + prob) - prob;      // straight-through value
        bool  sel   = mk > 0.5f;
        int slot = -1;
        if (sel) { slot = atomicAdd(counter, 1); idx[slot] = t; }
        maskv[t] = sel ? mk : 0.f;
        sh_rstd = rstd;
        sh_slot = slot;
    }
    __syncthreads();
    const float rstd = sh_rstd;
    const int   slot = sh_slot;
    float* outr = out + (size_t)t * C;
    __bf16* xnr = (slot >= 0) ? (xn + (size_t)slot * C) : nullptr;
    for (int i = tid; i < C; i += blockDim.x) {
        float v = xr[i];
        outr[i] = v;                              // pass-through residual
        if (slot >= 0) xnr[i] = (__bf16)(v * rstd);
    }
}

// ---------------------------------------------------------------------------
// Kernel 3: GEMM1  h = relu(xn @ w_fc^T)^2   (compacted rows only)
// ---------------------------------------------------------------------------
__global__ __launch_bounds__(256)
void gemm_fc_relu2_kernel(const __bf16* __restrict__ xn,
                          const __bf16* __restrict__ wfc,
                          __bf16* __restrict__ h,
                          const int* __restrict__ counter,
                          int K, int N) {
    const int count = *counter;
    const int rows0 = blockIdx.y * BM;
    if (rows0 >= count) return;
    const int cols0 = blockIdx.x * BM;

    v8f acc[4][2] = {};
    gemm_core(xn, wfc, K, rows0, cols0, acc);

    const int lane  = threadIdx.x & 31;
    const int wave  = threadIdx.x >> 5;
    const int rw = rows0 + (wave >> 2) * 64;
    const int cw = cols0 + (wave & 3) * 32;
    const int lr = lane & 15;
    const int rowHi = (lane >> 4) << 3;
#pragma unroll
    for (int mi = 0; mi < 4; ++mi) {
#pragma unroll
        for (int r = 0; r < 8; ++r) {
            int m = rw + mi * 16 + r + rowHi;
            if (m < count) {
#pragma unroll
                for (int ni = 0; ni < 2; ++ni) {
                    float v = acc[mi][ni][r];
                    v = fmaxf(v, 0.f);
                    h[(size_t)m * N + cw + ni * 16 + lr] = (__bf16)(v * v);
                }
            }
        }
    }
}

// ---------------------------------------------------------------------------
// Kernel 4: GEMM2  out[t] = x[t] + mask[t] * (h @ w_proj^T)   (scatter)
// ---------------------------------------------------------------------------
__global__ __launch_bounds__(256)
void gemm_proj_scatter_kernel(const __bf16* __restrict__ h,
                              const __bf16* __restrict__ wproj,
                              const float* __restrict__ x,
                              const int* __restrict__ idx,
                              const float* __restrict__ maskv,
                              float* __restrict__ out,
                              const int* __restrict__ counter,
                              int K, int N) {
    const int count = *counter;
    const int rows0 = blockIdx.y * BM;
    if (rows0 >= count) return;
    const int cols0 = blockIdx.x * BM;

    v8f acc[4][2] = {};
    gemm_core(h, wproj, K, rows0, cols0, acc);

    const int lane  = threadIdx.x & 31;
    const int wave  = threadIdx.x >> 5;
    const int rw = rows0 + (wave >> 2) * 64;
    const int cw = cols0 + (wave & 3) * 32;
    const int lr = lane & 15;
    const int rowHi = (lane >> 4) << 3;
#pragma unroll
    for (int mi = 0; mi < 4; ++mi) {
#pragma unroll
        for (int r = 0; r < 8; ++r) {
            int m = rw + mi * 16 + r + rowHi;
            if (m < count) {
                int   t  = idx[m];
                float mk = maskv[t];
                const float* xr   = x   + (size_t)t * N;
                float*       outr = out + (size_t)t * N;
#pragma unroll
                for (int ni = 0; ni < 2; ++ni) {
                    int n = cw + ni * 16 + lr;
                    outr[n] = xr[n] + mk * acc[mi][ni][r];
                }
            }
        }
    }
}

// ---------------------------------------------------------------------------
// Host-side launcher
// ---------------------------------------------------------------------------
extern "C" void kernel_launch(void* const* d_in, const int* in_sizes, int n_in,
                              void* d_out, int out_size, void* d_ws, size_t ws_size,
                              hipStream_t stream) {
    const float* x        = (const float*)d_in[0];   // (B,T,C)
    const float* w_fc     = (const float*)d_in[1];   // (F,C)
    const float* w_proj   = (const float*)d_in[2];   // (C,F)
    const float* w_router = (const float*)d_in[3];   // (1,C)

    const int C  = in_sizes[3];                      // 2048
    const int BT = in_sizes[0] / C;                  // 8192
    const int F  = in_sizes[1] / C;                  // 8192

    float* out = (float*)d_out;

    // Workspace layout (256-byte aligned regions)
    char* ws = (char*)d_ws;
    auto align256 = [](size_t o) { return (o + 255) & ~(size_t)255; };
    size_t off = 0;
    int*    counter = (int*)(ws + off);    off = align256(off + 4);
    int*    idx     = (int*)(ws + off);    off = align256(off + (size_t)BT * 4);
    float*  maskv   = (float*)(ws + off);  off = align256(off + (size_t)BT * 4);
    __bf16* wfcb    = (__bf16*)(ws + off); off = align256(off + (size_t)F * C * 2);
    __bf16* wprojb  = (__bf16*)(ws + off); off = align256(off + (size_t)F * C * 2);
    __bf16* xn      = (__bf16*)(ws + off); off = align256(off + (size_t)BT * C * 2);
    __bf16* hbuf    = (__bf16*)(ws + off); off = align256(off + (size_t)BT * F * 2);

    // 0) zero the compaction counter
    zero_counter_kernel<<<1, 32, 0, stream>>>(counter);

    // 1) convert weights to bf16
    {
        size_t n = (size_t)F * C;
        int blocks = (int)((n / 4 + 255) / 256);
        cvt_bf16_kernel<<<blocks, 256, 0, stream>>>(w_fc,   wfcb,   n);
        cvt_bf16_kernel<<<blocks, 256, 0, stream>>>(w_proj, wprojb, n);
    }

    // 2) router + rmsnorm + compaction + residual copy
    router_rms_kernel<<<BT, 256, 0, stream>>>(x, w_router, out, xn, idx,
                                              maskv, counter, C);

    // 3) GEMM1: h = relu(xn @ w_fc^T)^2 over compacted rows
    {
        dim3 grid(F / BM, BT / BM);
        gemm_fc_relu2_kernel<<<grid, 256, 0, stream>>>(xn, wfcb, hbuf,
                                                       counter, C, F);
    }

    // 4) GEMM2: out[t] = x[t] + mask[t] * (h @ w_proj^T), scattered
    {
        dim3 grid(C / BM, BT / BM);
        gemm_proj_scatter_kernel<<<grid, 256, 0, stream>>>(hbuf, wprojb, x,
                                                           idx, maskv, out,
                                                           counter, F, C);
    }
}
